// StackedLTC_28638841930235
// MI455X (gfx1250) — matrix operand order repaired and checked by reading, not compiled, not run on hardware
//
#include <hip/hip_runtime.h>
#include <stdint.h>

// ---------------------------------------------------------------------------
// StackedLTC scan for MI455X (gfx1250, wave32, WMMA)
//   BATCH=512, SEQ=1024, IN=64, HID=256, LAYERS=3, OUT=1
// ---------------------------------------------------------------------------

#define BATCHN 512
#define SEQN   1024
#define INN    64
#define HIDN   256

typedef __attribute__((ext_vector_type(16))) __bf16 v16bf;
typedef __attribute__((ext_vector_type(8)))  __bf16 v8bf;
typedef __attribute__((ext_vector_type(8)))  float  v8f;

// Global (addrspace 1) pointer flavors: keep weight loads on the VMEM/global
// path (LOADcnt only) instead of FLAT (LOADcnt+DScnt, arbitrates vs LDS pipe).
typedef __attribute__((address_space(1))) const __bf16 gbf16;
typedef __attribute__((address_space(1))) const v16bf gv16bf;

constexpr int    BT          = 64;                 // batch tile per workgroup
constexpr int    COMB_STRIDE = 520;                // 512 + 8 bf16 pad (bank-conflict-free b128)
constexpr int    H_STRIDE    = 260;                // 256 + 4 f32 pad
constexpr int    FAN0        = INN + HIDN;         // 320
constexpr int    FAN12       = 2 * HIDN;           // 512
constexpr int    KC0         = FAN0 / 32;          // 10 k-chunks
constexpr int    KC12        = FAN12 / 32;         // 16 k-chunks
// Pre-swizzled weight fragments in d_ws: [kchunk][ntile(16)][lane(32)][elem(16)] bf16
constexpr size_t FRAG_L0  = 0;
constexpr size_t FRAG_L1  = (size_t)KC0  * 16 * 32 * 16;     // 81920
constexpr size_t FRAG_L2  = FRAG_L1 + (size_t)KC12 * 16 * 32 * 16;
constexpr size_t FRAG_TOT = FRAG_L2 + (size_t)KC12 * 16 * 32 * 16;   // 344064 bf16
constexpr size_t SMEM_BYTES = (size_t)BT * COMB_STRIDE * 2     // bf16 A staging
                            + (size_t)3 * BT * H_STRIDE * 4;   // f32 hidden state

__device__ __forceinline__ float fast_tanh(float x) {
    x = fminf(fmaxf(x, -15.0f), 15.0f);
    float e = __expf(2.0f * x);
    return (e - 1.0f) * __builtin_amdgcn_rcpf(e + 1.0f);
}

// ---------------------------------------------------------------------------
// Weight swizzle: f32 row-major W[n][k]  ->  bf16 WMMA B-fragment layout.
// B-matrix (32x16 bf16, wave32): lane&15 = N, lane>>4 selects K half:
//   element e of lane -> K = kchunk*32 + 8*(lane>>4) + (e&7) + 16*(e>>3), N = ntile*16 + (lane&15)
// ---------------------------------------------------------------------------
__global__ void ltc_prep_weights(const float* __restrict__ W0,
                                 const float* __restrict__ W1,
                                 const float* __restrict__ W2,
                                 __bf16* __restrict__ frags) {
    size_t i = (size_t)blockIdx.x * blockDim.x + threadIdx.x;
    if (i >= FRAG_TOT) return;
    const float* W; size_t rel; int fan;
    if (i < FRAG_L1)      { W = W0; rel = i;            fan = FAN0;  }
    else if (i < FRAG_L2) { W = W1; rel = i - FRAG_L1;  fan = FAN12; }
    else                  { W = W2; rel = i - FRAG_L2;  fan = FAN12; }
    int e    = (int)(rel & 15);
    int lane = (int)((rel >> 4) & 31);
    int nt   = (int)((rel >> 9) & 15);
    int kc   = (int)(rel >> 13);
    int n = nt * 16 + (lane & 15);
    int k = kc * 32 + 8 * (lane >> 4) + (e & 7) + 16 * (e >> 3);
    frags[i] = (__bf16)W[(size_t)n * fan + k];
}

// ---------------------------------------------------------------------------
// Main sequential scan. Grid = BATCH/BT = 8 workgroups, 512 threads (16 waves).
// Wave w: M-group mg = w&1 (two 16-row tiles), N-group ng = w>>1 (two 16-col tiles).
// ---------------------------------------------------------------------------
__launch_bounds__(512, 1)
__global__ void ltc_scan_kernel(const float* __restrict__ x,
                                const __bf16* __restrict__ frags,
                                const float* __restrict__ bias0, const float* __restrict__ tc0,
                                const float* __restrict__ bias1, const float* __restrict__ tc1,
                                const float* __restrict__ bias2, const float* __restrict__ tc2,
                                const float* __restrict__ Wh,    const float* __restrict__ bh,
                                float* __restrict__ out) {
    extern __shared__ char smem[];
    __bf16* comb = (__bf16*)smem;                                     // [BT][COMB_STRIDE] bf16
    float*  hf   = (float*)(smem + (size_t)BT * COMB_STRIDE * 2);     // [3][BT][H_STRIDE] f32

    const int tid  = threadIdx.x;
    const int lane = tid & 31;
    const int wave = tid >> 5;
    const int half = lane >> 4;
    const int nloc = lane & 15;
    const int mg   = wave & 1;
    const int ng   = wave >> 1;
    const int bbase = blockIdx.x * BT;

    // zero the f32 hidden state
    for (int i = tid; i < 3 * BT * H_STRIDE; i += 512) hf[i] = 0.0f;

    // per-lane time-invariant bias / tau for the two owned N-tiles
    float bias[3][2], tau[3][2];
    {
        const float* bptr[3] = {bias0, bias1, bias2};
        const float* tptr[3] = {tc0, tc1, tc2};
        #pragma unroll
        for (int l = 0; l < 3; ++l)
            #pragma unroll
            for (int nt = 0; nt < 2; ++nt) {
                int n = ng * 32 + nt * 16 + nloc;
                bias[l][nt] = bptr[l][n];
                tau[l][nt]  = __builtin_amdgcn_rcpf(1.0f + __expf(-tptr[l][n]));
            }
    }
    __syncthreads();

    const int xr  = tid >> 3;      // 0..63: staging row
    const int xc8 = tid & 7;       // 8-float / 32-elem column group
    const float* xptr = x + (size_t)(bbase + xr) * SEQN * INN + xc8 * 8;

    // Opaque bits of the fragment base pointer. Redefined (no-op asm) every
    // time-step so LICM cannot hoist the weight loads out of the 1024-step
    // loop (avoids a 672-VGPR live range + scratch spills). Re-materialized
    // as an addrspace(1) pointer so loads lower to global_load_b128.
    uintptr_t frbits = (uintptr_t)frags;

    for (int t = 0; t < SEQN; ++t) {
        asm volatile("" : "+s"(frbits));   // keep B-fragment loads inside the time loop
        gbf16* fr = (gbf16*)frbits;

        // ---- stage x_t (bf16) into comb[:,0:64) and h0 into comb[:,64:320) ----
        {
            float4 a = *(const float4*)xptr;
            float4 b = *(const float4*)(xptr + 4);
            __bf16* c = comb + (size_t)xr * COMB_STRIDE + xc8 * 8;
            c[0]=(__bf16)a.x; c[1]=(__bf16)a.y; c[2]=(__bf16)a.z; c[3]=(__bf16)a.w;
            c[4]=(__bf16)b.x; c[5]=(__bf16)b.y; c[6]=(__bf16)b.z; c[7]=(__bf16)b.w;
            __builtin_prefetch(xptr + INN, 0, 1);   // global_prefetch_b8 for x_{t+1}
        }
        xptr += INN;
        {
            const float* hsrc = hf + (size_t)xr * H_STRIDE + xc8 * 32;
            __bf16* dst = comb + (size_t)xr * COMB_STRIDE + INN + xc8 * 32;
            #pragma unroll
            for (int c2 = 0; c2 < 32; ++c2) dst[c2] = (__bf16)hsrc[c2];
        }
        __syncthreads();

        #pragma unroll
        for (int l = 0; l < 3; ++l) {
            const int KC = (l == 0) ? KC0 : KC12;
            gbf16* wf = fr
                + (l == 0 ? FRAG_L0 : (l == 1 ? FRAG_L1 : FRAG_L2))
                + ((size_t)(2 * ng) * 32 + lane) * 16;

            v8f acc[2][2] = {};
            for (int kk = 0; kk < KC; ++kk) {
                const size_t ko = (size_t)kk * (16 * 32 * 16);
                v16bf B0 = *(gv16bf*)(wf + ko);            // N-tile 2ng
                v16bf B1 = *(gv16bf*)(wf + ko + 32 * 16);  // N-tile 2ng+1
                // A 16x32 bf16 lane map: elems 0..7 -> K = kb..kb+7,
                // elems 8..15 -> K = kb+16..kb+23, kb = kk*32 + 8*(lane>>4)
                const int kb = kk * 32 + 8 * half;
                union { v16bf v; v8bf h[2]; } A0, A1;
                const __bf16* r0 = comb + (size_t)((2 * mg + 0) * 16 + nloc) * COMB_STRIDE + kb;
                const __bf16* r1 = comb + (size_t)((2 * mg + 1) * 16 + nloc) * COMB_STRIDE + kb;
                A0.h[0] = *(const v8bf*)(r0);
                A0.h[1] = *(const v8bf*)(r0 + 16);
                A1.h[0] = *(const v8bf*)(r1);
                A1.h[1] = *(const v8bf*)(r1 + 16);
                acc[0][0] = __builtin_amdgcn_wmma_f32_16x16x32_bf16(false, A0.v, false, B0, (short)0, acc[0][0], false, false);
                acc[0][1] = __builtin_amdgcn_wmma_f32_16x16x32_bf16(false, A0.v, false, B1, (short)0, acc[0][1], false, false);
                acc[1][0] = __builtin_amdgcn_wmma_f32_16x16x32_bf16(false, A1.v, false, B0, (short)0, acc[1][0], false, false);
                acc[1][1] = __builtin_amdgcn_wmma_f32_16x16x32_bf16(false, A1.v, false, B1, (short)0, acc[1][1], false, false);
            }

            // tanh + tau blend, keep f32 master state in LDS
            float hnew[2][2][8];
            #pragma unroll
            for (int mt = 0; mt < 2; ++mt)
                #pragma unroll
                for (int nt = 0; nt < 2; ++nt)
                    #pragma unroll
                    for (int r = 0; r < 8; ++r) {
                        // C/D layout: VGPR r -> M = r + 8*half; N = lane&15
                        int m = (2 * mg + mt) * 16 + r + 8 * half;
                        int n = ng * 32 + nt * 16 + nloc;
                        float* hp = hf + (size_t)(l * BT + m) * H_STRIDE + n;
                        float nh = fast_tanh(acc[mt][nt][r] + bias[l][nt]);
                        float ho = *hp;
                        float hv = ho + tau[l][nt] * (nh - ho);
                        *hp = hv;
                        hnew[mt][nt][r] = hv;
                    }
            __syncthreads();   // all waves done reading comb(A_l); hf[l] fully updated

            if (l < 2) {
                // new h_l (bf16) -> comb[:,0:256); old h_{l+1} (bf16) -> comb[:,256:512)
                #pragma unroll
                for (int mt = 0; mt < 2; ++mt)
                    #pragma unroll
                    for (int nt = 0; nt < 2; ++nt)
                        #pragma unroll
                        for (int r = 0; r < 8; ++r) {
                            int m = (2 * mg + mt) * 16 + r + 8 * half;
                            int n = ng * 32 + nt * 16 + nloc;
                            comb[(size_t)m * COMB_STRIDE + n] = (__bf16)hnew[mt][nt][r];
                        }
                const float* hsrc = hf + (size_t)((l + 1) * BT + xr) * H_STRIDE + xc8 * 32;
                __bf16* dst = comb + (size_t)xr * COMB_STRIDE + HIDN + xc8 * 32;
                #pragma unroll
                for (int c2 = 0; c2 < 32; ++c2) dst[c2] = (__bf16)hsrc[c2];
                __syncthreads();
            }
        }
        __syncthreads();   // layer-2 comb reads done before next step's staging
    }

    // ---- head: out[b] = h2[b,:] . Wh + bh  (OUT=1), exact f32 ----
    if (tid < BT) {
        const float* h2 = hf + (size_t)(2 * BT + tid) * H_STRIDE;
        float s = bh[0];
        for (int k = 0; k < HIDN; ++k) s += h2[k] * Wh[k];
        out[bbase + tid] = s;
    }
}

// ---------------------------------------------------------------------------
extern "C" void kernel_launch(void* const* d_in, const int* in_sizes, int n_in,
                              void* d_out, int out_size, void* d_ws, size_t ws_size,
                              hipStream_t stream) {
    const float* x   = (const float*)d_in[0];
    const float* W0  = (const float*)d_in[1];
    const float* b0  = (const float*)d_in[2];
    const float* tc0 = (const float*)d_in[3];
    const float* W1  = (const float*)d_in[4];
    const float* b1  = (const float*)d_in[5];
    const float* tc1 = (const float*)d_in[6];
    const float* W2  = (const float*)d_in[7];
    const float* b2  = (const float*)d_in[8];
    const float* tc2 = (const float*)d_in[9];
    const float* Wh  = (const float*)d_in[10];
    const float* bh  = (const float*)d_in[11];
    float* out = (float*)d_out;
    __bf16* frags = (__bf16*)d_ws;   // needs FRAG_TOT*2 = ~672 KB of workspace

    // Swizzle weights into WMMA B-fragment layout (runs every call; deterministic).
    {
        int total  = (int)FRAG_TOT;
        int blocks = (total + 255) / 256;
        ltc_prep_weights<<<blocks, 256, 0, stream>>>(W0, W1, W2, frags);
    }

    (void)hipFuncSetAttribute((const void*)ltc_scan_kernel,
                              hipFuncAttributeMaxDynamicSharedMemorySize,
                              (int)SMEM_BYTES);
    ltc_scan_kernel<<<BATCHN / BT, 512, SMEM_BYTES, stream>>>(
        x, frags, b0, tc0, b1, tc1, b2, tc2, Wh, bh, out);
}
